// PinSAGE_35124242547107
// MI455X (gfx1250) — matrix operand order repaired and checked by reading, not compiled, not run on hardware
//
#include <hip/hip_runtime.h>
#include <hip/hip_bf16.h>

// ---------------------------------------------------------------------------
// PinSAGE 2-layer forward, CDNA5 (gfx1250) wave32 + v_wmma_f32_16x16x32_bf16.
// Intermediates stored bf16 (t1 = 102MB -> mostly L2-resident for the gather),
// accumulation in f32 via WMMA.  Conversions use native bf16 casts so the
// backend emits hardware cvt ops instead of bit-twiddling VALU chains.
// ---------------------------------------------------------------------------

typedef __attribute__((ext_vector_type(16))) __bf16 v16bf;
typedef __attribute__((ext_vector_type(8)))  float  v8f;

static __device__ __forceinline__ float bf2f(__bf16 h) {
    return (float)h;                 // native bf16 -> f32
}

static __device__ __forceinline__ __bf16 f2bf(float f) {
    return (__bf16)f;                // native fptrunc, round-to-nearest-even
}

// ---------------------------------------------------------------------------
// Weight prep: src f32 [K][N] row-major -> dst bf16 [N][K] (B-fragment rows
// become contiguous 32B v16bf loads).
// ---------------------------------------------------------------------------
__global__ __launch_bounds__(256) void k_prep(const float* __restrict__ src,
                                              __bf16* __restrict__ dst,
                                              int K, int N) {
    int i = blockIdx.x * 256 + threadIdx.x;
    if (i >= K * N) return;
    int k = i / N, n = i % N;
    dst[n * K + k] = f2bf(src[k * N + n]);
}

// ---------------------------------------------------------------------------
// t = relu(X @ Wa + ba).  X: [Nrows][128] (f32 or bf16), Wt: bf16 [128][128]
// transposed ([n][k]), out Y: bf16 [Nrows][128].
// Block = 256 threads = 8 waves; wave w does rows blockIdx*128 + w*16 .. +15.
// ---------------------------------------------------------------------------
template <bool SRCF32>
__global__ __launch_bounds__(256) void k_transform(const void* __restrict__ Xv,
                                                   const __bf16* __restrict__ Wt,
                                                   const float* __restrict__ bias,
                                                   __bf16* __restrict__ Y,
                                                   int Nrows) {
    const int tid  = threadIdx.x;
    const int wave = tid >> 5;
    const int lane = tid & 31;
    const int half = lane >> 4;    // 0: lanes 0-15, 1: lanes 16-31
    const int l16  = lane & 15;
    const int rowTile = blockIdx.x * 128 + wave * 16;

    v8f acc[8];
    const v8f vz = {0.f, 0.f, 0.f, 0.f, 0.f, 0.f, 0.f, 0.f};
#pragma unroll
    for (int i = 0; i < 8; ++i) acc[i] = vz;

    int arow = rowTile + l16;                 // A-matrix row M for this lane
    if (arow > Nrows - 1) arow = Nrows - 1;   // clamp loads; stores guarded

    const float*  Xf = (const float*)Xv;
    const __bf16* Xh = (const __bf16*)Xv;

#pragma unroll
    for (int kb = 0; kb < 128; kb += 32) {
        // A fragment, contiguous K order (matches B):
        //   lanes 0-15 : elems 0-15 = K kb..kb+15
        //   lanes 16-31: elems 0-15 = K kb+16..kb+31
        v16bf a;
        const int k0 = kb + half * 16;
        const int k1 = kb + half * 16 + 8;
        if (SRCF32) {
            const float* r = Xf + arow * 128;
#pragma unroll
            for (int j = 0; j < 8; ++j) {
                a[j]     = f2bf(r[k0 + j]);
                a[8 + j] = f2bf(r[k1 + j]);
            }
        } else {
            const __bf16* r = Xh + arow * 128;
#pragma unroll
            for (int j = 0; j < 8; ++j) {
                a[j]     = r[k0 + j];
                a[8 + j] = r[k1 + j];
            }
        }
        // B fragments: col = nt*16 + l16; K = kb + half*16 + e (contig 16 bf16)
        const int kbase = kb + half * 16;
#pragma unroll
        for (int nt = 0; nt < 8; ++nt) {
            const int col = nt * 16 + l16;
            const v16bf b = *reinterpret_cast<const v16bf*>(Wt + col * 128 + kbase);
            acc[nt] = __builtin_amdgcn_wmma_f32_16x16x32_bf16(
                false, a, false, b, (short)0, acc[nt], false, false);
        }
    }

    // C/D layout: elem r in lane L -> row = r + (L>=16 ? 8:0), col = L&15
#pragma unroll
    for (int r = 0; r < 8; ++r) {
        const int orow = rowTile + half * 8 + r;
        if (orow < Nrows) {
#pragma unroll
            for (int nt = 0; nt < 8; ++nt) {
                const int col = nt * 16 + l16;
                float v = acc[nt][r] + bias[col];
                Y[orow * 128 + col] = f2bf(v > 0.f ? v : 0.f);
            }
        }
    }
}

// ---------------------------------------------------------------------------
// Layer aggregate + encode for 16 nodes per block.
// Phase 1: 256 threads build cat = [self | weighted-mean(neigh)] (16x256 bf16)
//          in LDS.  Phase 2: wave w = one 16-col output tile, K=256, 8 WMMAs.
// ---------------------------------------------------------------------------
template <bool SELF_F32, bool OUT_F32>
__global__ __launch_bounds__(256) void k_aggenc(const void* __restrict__ selfSrc,
                                                const __bf16* __restrict__ t,
                                                const int* __restrict__ nidx,
                                                const float* __restrict__ nw,
                                                const int* __restrict__ sidx,
                                                const __bf16* __restrict__ Wet,  // [128][256]
                                                const float* __restrict__ be,
                                                void* __restrict__ out,
                                                int N) {
    __shared__ __align__(32) __bf16 catA[16][256];
    __shared__ int   sIdx[16][16];
    __shared__ float sW[16][16];

    const int tid      = threadIdx.x;
    const int node     = tid >> 4;      // 0..15 (phase-1 row)
    const int l16n     = tid & 15;
    const int nodeBase = blockIdx.x * 16;
    int n = nodeBase + node;
    if (n > N - 1) n = N - 1;

    sIdx[node][l16n] = nidx[n * 16 + l16n];
    sW[node][l16n]   = nw[n * 16 + l16n];
    __syncthreads();

    float wsum = 0.f;
#pragma unroll
    for (int f = 0; f < 16; ++f) wsum += sW[node][f];
    const float inv = 1.f / wsum;

    // each thread owns 8 dims of its node
    const int d0 = l16n * 8;
    float acc[8];
#pragma unroll
    for (int j = 0; j < 8; ++j) acc[j] = 0.f;

    for (int f = 0; f < 16; ++f) {
        const float w    = sW[node][f];
        const __bf16* r  = t + (size_t)sIdx[node][f] * 128 + d0;
#pragma unroll
        for (int j = 0; j < 8; ++j) acc[j] += w * bf2f(r[j]);
    }

    const int si = sidx[n];
    if (SELF_F32) {
        const float* sr = (const float*)selfSrc + (size_t)si * 128 + d0;
#pragma unroll
        for (int j = 0; j < 8; ++j) catA[node][d0 + j] = f2bf(sr[j]);
    } else {
        const __bf16* sr = (const __bf16*)selfSrc + (size_t)si * 128 + d0;
#pragma unroll
        for (int j = 0; j < 8; ++j) catA[node][d0 + j] = sr[j];
    }
#pragma unroll
    for (int j = 0; j < 8; ++j) catA[node][128 + d0 + j] = f2bf(acc[j] * inv);
    __syncthreads();

    // ---- Phase 2: encoder GEMM (16 rows x 256 K) x (256 x 128) ----
    const int lane = tid & 31;
    const int wave = tid >> 5;      // N-tile
    const int half = lane >> 4;
    const int l16  = lane & 15;
    const int col  = wave * 16 + l16;

    v8f c = {0.f, 0.f, 0.f, 0.f, 0.f, 0.f, 0.f, 0.f};
#pragma unroll
    for (int kb = 0; kb < 256; kb += 32) {
        v16bf a;
        const int k0 = kb + half * 16;
        const int k1 = kb + half * 16 + 8;
        const __bf16* ar = &catA[l16][0];
#pragma unroll
        for (int j = 0; j < 8; ++j) {
            a[j]     = ar[k0 + j];
            a[8 + j] = ar[k1 + j];
        }
        const v16bf b = *reinterpret_cast<const v16bf*>(Wet + col * 256 + kb + half * 16);
        c = __builtin_amdgcn_wmma_f32_16x16x32_bf16(
            false, a, false, b, (short)0, c, false, false);
    }

    const float bcol = be[col];
#pragma unroll
    for (int r = 0; r < 8; ++r) {
        const int orow = nodeBase + half * 8 + r;
        if (orow < N) {
            float v = c[r] + bcol;
            v = v > 0.f ? v : 0.f;
            if (OUT_F32) ((float*)out)[orow * 128 + col] = v;
            else         ((__bf16*)out)[orow * 128 + col] = f2bf(v);
        }
    }
}

// ---------------------------------------------------------------------------
extern "C" void kernel_launch(void* const* d_in, const int* in_sizes, int n_in,
                              void* d_out, int out_size, void* d_ws, size_t ws_size,
                              hipStream_t stream) {
    const float* emb   = (const float*)d_in[0];
    const int*   nidx1 = (const int*)  d_in[1];
    const float* nw1   = (const float*)d_in[2];
    const int*   sidx1 = (const int*)  d_in[3];
    const int*   nidx2 = (const int*)  d_in[4];
    const float* nw2   = (const float*)d_in[5];
    const int*   sidx2 = (const int*)  d_in[6];
    const float* Wa1   = (const float*)d_in[7];
    const float* ba1   = (const float*)d_in[8];
    const float* We1   = (const float*)d_in[9];
    const float* be1   = (const float*)d_in[10];
    const float* Wa2   = (const float*)d_in[11];
    const float* ba2   = (const float*)d_in[12];
    const float* We2   = (const float*)d_in[13];
    const float* be2   = (const float*)d_in[14];

    const int N0 = in_sizes[0] / 128;   // 400000
    const int N1 = in_sizes[3];         // 100000
    const int N2 = in_sizes[6];         // 20000

    // workspace layout (bf16), all offsets 256B-aligned
    char* ws = (char*)d_ws;
    __bf16* Wa1t = (__bf16*)(ws + 0);        // 128*128*2 = 32KB
    __bf16* We1t = (__bf16*)(ws + 32768);    // 128*256*2 = 64KB
    __bf16* Wa2t = (__bf16*)(ws + 98304);    // 32KB
    __bf16* We2t = (__bf16*)(ws + 131072);   // 64KB
    __bf16* t1   = (__bf16*)(ws + 196608);   // N0*128*2 = 102.4MB
    __bf16* h1   = t1 + (size_t)N0 * 128;    // N1*128*2 = 25.6MB
    __bf16* t2   = h1 + (size_t)N1 * 128;    // N1*128*2 = 25.6MB

    // weight transpose + bf16 convert
    k_prep<<<(128 * 128 + 255) / 256, 256, 0, stream>>>(Wa1, Wa1t, 128, 128);
    k_prep<<<(256 * 128 + 255) / 256, 256, 0, stream>>>(We1, We1t, 256, 128);
    k_prep<<<(128 * 128 + 255) / 256, 256, 0, stream>>>(Wa2, Wa2t, 128, 128);
    k_prep<<<(256 * 128 + 255) / 256, 256, 0, stream>>>(We2, We2t, 256, 128);

    // Layer 1
    k_transform<true><<<(N0 + 127) / 128, 256, 0, stream>>>(emb, Wa1t, ba1, t1, N0);
    k_aggenc<true, false><<<(N1 + 15) / 16, 256, 0, stream>>>(
        emb, t1, nidx1, nw1, sidx1, We1t, be1, h1, N1);

    // Layer 2
    k_transform<false><<<(N1 + 127) / 128, 256, 0, stream>>>(h1, Wa2t, ba2, t2, N1);
    k_aggenc<false, true><<<(N2 + 15) / 16, 256, 0, stream>>>(
        h1, t2, nidx2, nw2, sidx2, We2t, be2, d_out, N2);
}
